// SwinTransformerBlock_with_DMlp_46823733461566
// MI455X (gfx1250) — matrix operand order repaired and checked by reading, not compiled, not run on hardware
//
#include <hip/hip_runtime.h>
#include <cstdint>
#include <cstddef>

typedef _Float16 f16_t;
typedef __attribute__((ext_vector_type(16))) _Float16 v16h;
typedef __attribute__((ext_vector_type(8)))  _Float16 v8h;
typedef __attribute__((ext_vector_type(8)))  float    v8f;

// ---- static config ----
#define BB   8
#define HH   56
#define WWD  56
#define CC   576
#define NN   (HH*WWD)        // 3136
#define NTOK (BB*NN)         // 25088
#define NHD  6
#define HD   96
#define NWIN 512
#define INF  1600
#define HIDF 1024

// 25 masked kernel offsets (row-major over the 7x7 mask)
__constant__ int c_di[25] = {0,0,0, 1,1,1, 2,2,2, 3,3,3,3,3,3,3, 4,4,4, 5,5,5, 6,6,6};
__constant__ int c_dj[25] = {0,3,6, 1,3,5, 2,3,4, 0,1,2,3,4,5,6, 2,3,4, 1,3,5, 0,3,6};

// ---------------- WMMA fragment helpers (ISA 7.12.2 layouts) ----------------
// A 16x32 f16 fragment: per lane, two contiguous 8-half runs -> two b128 loads.
__device__ __forceinline__ v16h load_a_frag(const f16_t* __restrict__ A, int lda, int row0, int k0) {
  const int lane = threadIdx.x & 31;
  const int hf   = lane >> 4;
  const int m    = lane & 15;
  const f16_t* p = A + (size_t)(row0 + m) * lda + k0 + hf * 8;
  v8h lo = *(const v8h*)(p);
  v8h hi = *(const v8h*)(p + 16);
  return __builtin_shufflevector(lo, hi, 0,1,2,3,4,5,6,7,8,9,10,11,12,13,14,15);
}

// B fragments come from a pre-swizzled (fragment-major) weight buffer:
// tile (kt,nt) of 32x16 stored as 512 contiguous f16, lane-major (16 per lane).
__device__ __forceinline__ v16h load_b_packed(const f16_t* __restrict__ Bp, int Ktiles, int k0, int n0) {
  const int lane = threadIdx.x & 31;
  const f16_t* p = Bp + (((size_t)(n0 >> 4) * Ktiles + (k0 >> 5)) << 9) + lane * 16;
  return *(const v16h*)p;
}

__device__ __forceinline__ v8f wmma_f16(v16h a, v16h b, v8f c) {
  return __builtin_amdgcn_wmma_f32_16x16x32_f16(false, a, false, b, (short)0, c, false, false);
}

__device__ __forceinline__ float gelu_exact(float x) {
  return 0.5f * x * (1.0f + erff(x * 0.70710678118654752f));
}

// ---------------- generic tiled WMMA GEMM ----------------
// Block: 256 threads = 8 waves (4 along M x 2 along N). Wave tile 64x32 -> 8 WMMA/K-step.
// Block tile: 256(M) x 64(N). All problem dims divide exactly; EXEC stays all-ones.
// EPI: 0 = f16 out (+bias); 1 = f16 out (+bias, exact GELU); 2 = f32 out (+bias, +residual)
template <int EPI>
__global__ __launch_bounds__(256) void gemm_wmma_kernel(
    const f16_t* __restrict__ A, const f16_t* __restrict__ Bp,
    const float* __restrict__ bias, const float* __restrict__ resid,
    void* __restrict__ Dout, int N, int K, int lda, int ldd)
{
  const int wave = threadIdx.x >> 5;
  const int wrow = wave & 3;              // 4 waves along M
  const int wcol = wave >> 2;             // 2 waves along N
  const int m0 = blockIdx.x * 256 + wrow * 64;
  const int n0 = blockIdx.y * 64  + wcol * 32;
  const int Ktiles = K >> 5;

  v8f acc[8];
#pragma unroll
  for (int i = 0; i < 8; ++i) acc[i] = (v8f){};

  for (int k0 = 0; k0 < K; k0 += 32) {
    v16h b0 = load_b_packed(Bp, Ktiles, k0, n0);
    v16h b1 = load_b_packed(Bp, Ktiles, k0, n0 + 16);
    v16h a0 = load_a_frag(A, lda, m0,      k0);
    v16h a1 = load_a_frag(A, lda, m0 + 16, k0);
    v16h a2 = load_a_frag(A, lda, m0 + 32, k0);
    v16h a3 = load_a_frag(A, lda, m0 + 48, k0);
    acc[0] = wmma_f16(a0, b0, acc[0]);
    acc[1] = wmma_f16(a0, b1, acc[1]);
    acc[2] = wmma_f16(a1, b0, acc[2]);
    acc[3] = wmma_f16(a1, b1, acc[3]);
    acc[4] = wmma_f16(a2, b0, acc[4]);
    acc[5] = wmma_f16(a2, b1, acc[5]);
    acc[6] = wmma_f16(a3, b0, acc[6]);
    acc[7] = wmma_f16(a3, b1, acc[7]);
  }

  const int lane = threadIdx.x & 31;
  const int hf   = lane >> 4;
  const int nn   = lane & 15;
#pragma unroll
  for (int mi = 0; mi < 4; ++mi) {
#pragma unroll
    for (int ni = 0; ni < 2; ++ni) {
      const v8f c   = acc[mi * 2 + ni];
      const int rb  = m0 + mi * 16 + hf * 8;
      const int cb  = n0 + ni * 16 + nn;
      const float bv = bias[cb];
#pragma unroll
      for (int r = 0; r < 8; ++r) {
        const size_t idx = (size_t)(rb + r) * ldd + cb;
        float v = c[r] + bv;
        if (EPI == 0) {
          ((f16_t*)Dout)[idx] = (f16_t)v;
        } else if (EPI == 1) {
          ((f16_t*)Dout)[idx] = (f16_t)gelu_exact(v);
        } else {
          ((float*)Dout)[idx] = v + resid[idx];
        }
      }
    }
  }
}

// ---------------- weight pack: f32 [KxN] -> f16 B-fragment-major tiles ----------------
__global__ void pack_b_kernel(const float* __restrict__ src, f16_t* __restrict__ dst, int K, int N)
{
  const int idx = blockIdx.x * 256 + threadIdx.x;
  if (idx >= K * N) return;
  const int ptile = idx >> 9;
  const int rem   = idx & 511;
  const int lane  = rem >> 4;
  const int i     = rem & 15;
  const int Ktiles = K >> 5;
  const int nt = ptile / Ktiles;
  const int kt = ptile % Ktiles;
  const int k = kt * 32 + (lane >> 4) * 8 + (i & 7) + (i >> 3) * 16;
  const int n = nt * 16 + (lane & 15);
  dst[idx] = (f16_t)src[(size_t)k * N + n];
}

// ---------------- layernorm (one token row per block) ----------------
__global__ __launch_bounds__(256) void ln_kernel(
    const float* __restrict__ X, const float* __restrict__ w, const float* __restrict__ b,
    f16_t* __restrict__ Y, int window_order)
{
  __shared__ float red[256];
  const int row = blockIdx.x;                 // token row: b*3136 + h*56 + w
  const float* xr = X + (size_t)row * CC;
  const int tid = threadIdx.x;

  float s = 0.f;
  for (int c = tid; c < CC; c += 256) s += xr[c];
  red[tid] = s; __syncthreads();
  for (int off = 128; off > 0; off >>= 1) { if (tid < off) red[tid] += red[tid + off]; __syncthreads(); }
  const float mean = red[0] / (float)CC;
  __syncthreads();

  float v = 0.f;
  for (int c = tid; c < CC; c += 256) { float d = xr[c] - mean; v += d * d; }
  red[tid] = v; __syncthreads();
  for (int off = 128; off > 0; off >>= 1) { if (tid < off) red[tid] += red[tid + off]; __syncthreads(); }
  const float rstd = rsqrtf(red[0] / (float)CC + 1e-5f);

  size_t orow = (size_t)row;
  if (window_order) {
    const int bb = row / NN, n = row % NN, h = n / WWD, ww = n % WWD;
    orow = (size_t)((bb * 64 + (h / 7) * 8 + (ww / 7)) * 49 + (h % 7) * 7 + (ww % 7));
  }
  f16_t* yr = Y + orow * CC;
  for (int c = tid; c < CC; c += 256)
    yr[c] = (f16_t)((xr[c] - mean) * rstd * w[c] + b[c]);
}

// ---------------- window attention (fp32; ~3% of FLOPs) ----------------
__global__ __launch_bounds__(64) void attn_kernel(
    const f16_t* __restrict__ qkv, const float* __restrict__ rpb, f16_t* __restrict__ o)
{
  const int blk = blockIdx.x;             // 0..3071
  const int win = blk / NHD, h = blk % NHD;
  const int t = threadIdx.x;
  if (t >= 49) return;

  const size_t base = (size_t)win * 49 * (3 * CC);
  const f16_t* qp = qkv + base + (size_t)t * (3 * CC) + h * HD;
  const float scale = 0.10206207261596575f;   // 96^-0.5
  float q[HD];
#pragma unroll
  for (int d = 0; d < HD; ++d) q[d] = (float)qp[d] * scale;

  float s[49];
  const int ty = t / 7, tx = t % 7;
  float mx = -3.0e38f;
  for (int j = 0; j < 49; ++j) {
    const f16_t* kp = qkv + base + (size_t)j * (3 * CC) + CC + h * HD;
    float acc = 0.f;
#pragma unroll 8
    for (int d = 0; d < HD; ++d) acc += q[d] * (float)kp[d];
    const int uy = j / 7, ux = j % 7;
    acc += rpb[((ty - uy + 6) * 13 + (tx - ux + 6)) * NHD + h];
    s[j] = acc;
    mx = fmaxf(mx, acc);
  }
  float sum = 0.f;
  for (int j = 0; j < 49; ++j) { s[j] = expf(s[j] - mx); sum += s[j]; }
  const float inv = 1.f / sum;

  float acc[HD];
#pragma unroll
  for (int d = 0; d < HD; ++d) acc[d] = 0.f;
  for (int j = 0; j < 49; ++j) {
    const float p = s[j] * inv;
    const f16_t* vp = qkv + base + (size_t)j * (3 * CC) + 2 * CC + h * HD;
#pragma unroll 8
    for (int d = 0; d < HD; ++d) acc[d] += p * (float)vp[d];
  }
  f16_t* op = o + ((size_t)win * 49 + t) * CC + h * HD;
#pragma unroll
  for (int d = 0; d < HD; ++d) op[d] = (f16_t)acc[d];
}

// ---------------- window reverse + residual add ----------------
__global__ void unwindow_residual(const float* __restrict__ X, const f16_t* __restrict__ O,
                                  float* __restrict__ X1)
{
  const size_t i = (size_t)blockIdx.x * blockDim.x + threadIdx.x;
  if (i >= (size_t)NTOK * CC) return;
  const int c = (int)(i % CC);
  const int row = (int)(i / CC);
  const int bb = row / NN, n = row % NN, h = n / WWD, ww = n % WWD;
  const size_t wrow = (size_t)((bb * 64 + (h / 7) * 8 + (ww / 7)) * 49 + (h % 7) * 7 + (ww % 7));
  X1[i] = X[i] + (float)O[wrow * CC + c];
}

// ---------------- DMlp gather: pixelshuffle(3) + reflectpad(2) + masked unfold ----------------
__device__ __forceinline__ int refl168(int t) {
  if (t < 0) t = -t;
  if (t >= 168) t = 334 - t;   // 2*168-2 - t
  return t;
}

__global__ void dmlp_gather(const f16_t* __restrict__ xn2, f16_t* __restrict__ cols)
{
  const size_t i = (size_t)blockIdx.x * blockDim.x + threadIdx.x;
  if (i >= (size_t)NTOK * INF) return;
  const int f   = (int)(i % INF);
  const int row = (int)(i / INF);
  const int nf = f / 25, p = f % 25;
  const int bb = row / NN, n = row % NN, h = n / WWD, ww = n % WWD;
  const int Y = refl168(3 * h  + c_di[p] - 2);
  const int X = refl168(3 * ww + c_dj[p] - 2);
  const int hp = Y / 3, r = Y % 3, wp = X / 3, s = X % 3;
  cols[i] = xn2[((size_t)bb * NN + hp * WWD + wp) * CC + nf * 9 + r * 3 + s];
}

// ---------------- host orchestration ----------------
extern "C" void kernel_launch(void* const* d_in, const int* in_sizes, int n_in,
                              void* d_out, int out_size, void* d_ws, size_t ws_size,
                              hipStream_t stream)
{
  (void)in_sizes; (void)n_in; (void)out_size; (void)ws_size;
  const float* x       = (const float*)d_in[0];
  const float* norm1_w = (const float*)d_in[1];
  const float* norm1_b = (const float*)d_in[2];
  const float* qkv_w   = (const float*)d_in[3];
  const float* qkv_b   = (const float*)d_in[4];
  const float* rpb     = (const float*)d_in[5];
  const float* proj_w  = (const float*)d_in[6];
  const float* proj_b  = (const float*)d_in[7];
  const float* norm2_w = (const float*)d_in[8];
  const float* norm2_b = (const float*)d_in[9];
  const float* fc1_w   = (const float*)d_in[10];
  const float* fc1_b   = (const float*)d_in[11];
  const float* fc2_w   = (const float*)d_in[12];
  const float* fc2_b   = (const float*)d_in[13];
  float* out = (float*)d_out;

  // workspace layout (slots reused across phases)
  char* ws = (char*)d_ws;
  size_t off = 0;
  auto take = [&](size_t bytes) { size_t o = off; off += (bytes + 255) & ~(size_t)255; return o; };
  f16_t* slotA   = (f16_t*)(ws + take((size_t)NTOK * CC   * 2)); // xn_win16 -> oproj16 -> xn2_16
  f16_t* slotB   = (f16_t*)(ws + take((size_t)NTOK * 3*CC * 2)); // qkv16 -> cols16
  f16_t* slotC   = (f16_t*)(ws + take((size_t)NTOK * HIDF * 2)); // o_win16 -> h16
  float* x1      = (float*)(ws + take((size_t)NTOK * CC   * 4));
  f16_t* qkv_wp  = (f16_t*)(ws + take((size_t)CC * 3*CC   * 2));
  f16_t* proj_wp = (f16_t*)(ws + take((size_t)CC * CC     * 2));
  f16_t* fc1_wp  = (f16_t*)(ws + take((size_t)INF * HIDF  * 2));
  f16_t* fc2_wp  = (f16_t*)(ws + take((size_t)HIDF * CC   * 2));

  // 1) pack weights into f16 B-fragment-major tiles
  {
    int n;
    n = CC * 3 * CC;  pack_b_kernel<<<(n + 255) / 256, 256, 0, stream>>>(qkv_w,  qkv_wp,  CC,   3 * CC);
    n = CC * CC;      pack_b_kernel<<<(n + 255) / 256, 256, 0, stream>>>(proj_w, proj_wp, CC,   CC);
    n = INF * HIDF;   pack_b_kernel<<<(n + 255) / 256, 256, 0, stream>>>(fc1_w,  fc1_wp,  INF,  HIDF);
    n = HIDF * CC;    pack_b_kernel<<<(n + 255) / 256, 256, 0, stream>>>(fc2_w,  fc2_wp,  HIDF, CC);
  }

  // 2) LN1 + window partition -> slotA (window-ordered f16)
  ln_kernel<<<NTOK, 256, 0, stream>>>(x, norm1_w, norm1_b, slotA, 1);

  // 3) QKV GEMM: [25088 x 576] x [576 x 1728] -> slotB (f16, +bias)
  gemm_wmma_kernel<0><<<dim3(NTOK / 256, (3 * CC) / 64), 256, 0, stream>>>(
      slotA, qkv_wp, qkv_b, nullptr, slotB, 3 * CC, CC, CC, 3 * CC);

  // 4) window attention -> slotC (o_win f16)
  attn_kernel<<<NWIN * NHD, 64, 0, stream>>>(slotB, rpb, slotC);

  // 5) proj GEMM: [25088 x 576] x [576 x 576] -> slotA (f16, +bias)
  gemm_wmma_kernel<0><<<dim3(NTOK / 256, CC / 64), 256, 0, stream>>>(
      slotC, proj_wp, proj_b, nullptr, slotA, CC, CC, CC, CC);

  // 6) window reverse + residual: x1 = x + unwindow(slotA)
  {
    size_t n = (size_t)NTOK * CC;
    unwindow_residual<<<(unsigned)((n + 255) / 256), 256, 0, stream>>>(x, slotA, x1);
  }

  // 7) LN2 (token order) -> slotA (f16)
  ln_kernel<<<NTOK, 256, 0, stream>>>(x1, norm2_w, norm2_b, slotA, 0);

  // 8) DMlp gather -> slotB (cols f16, [25088 x 1600])
  {
    size_t n = (size_t)NTOK * INF;
    dmlp_gather<<<(unsigned)((n + 255) / 256), 256, 0, stream>>>(slotA, slotB);
  }

  // 9) fc1 GEMM + GELU: [25088 x 1600] x [1600 x 1024] -> slotC (f16)
  gemm_wmma_kernel<1><<<dim3(NTOK / 256, HIDF / 64), 256, 0, stream>>>(
      slotB, fc1_wp, fc1_b, nullptr, slotC, HIDF, INF, INF, HIDF);

  // 10) fc2 GEMM + bias + residual: [25088 x 1024] x [1024 x 576] + x1 -> out (f32)
  gemm_wmma_kernel<2><<<dim3(NTOK / 256, CC / 64), 256, 0, stream>>>(
      slotC, fc2_wp, fc2_b, x1, out, CC, HIDF, HIDF, CC);
}
